// rnn_decoder_72103910965396
// MI455X (gfx1250) — matrix operand order repaired and checked by reading, not compiled, not run on hardware
//
#include <hip/hip_runtime.h>
#include <hip/hip_bf16.h>
#include <math.h>

// ---------------- problem dims ----------------
#define Bq    16
#define Hh    24
#define Ww    96
#define Cc    1024
#define HIDN  256
#define Vv    5000
#define EMBN  256
#define HWn   (Hh*Ww)          // 2304
#define BHWn  (Bq*HWn)         // 36864
#define BN_EPS 1e-5f

typedef __attribute__((ext_vector_type(16))) __bf16 v16bf;
typedef __attribute__((ext_vector_type(8)))  __bf16 v8bf;
typedef __attribute__((ext_vector_type(8)))  float  v8f;

// ---------------- WMMA fragment loads (contiguous k = 16*(lane/16) + s in A and B, wave32) --------
// A tile 16x32 row-major (row stride 32) in LDS, 64B-aligned rows.
__device__ __forceinline__ v16bf frag_a(const __bf16* A, int lane) {
  const int row = lane & 15;
  const int kb  = (lane >> 4) << 4;
  const __bf16* p = A + row * 32 + kb;
  v8bf lo = *(const v8bf*)(p);
  v8bf hi = *(const v8bf*)(p + 8);
  return __builtin_shufflevector(lo, hi, 0, 1, 2, 3, 4, 5, 6, 7,
                                         8, 9, 10, 11, 12, 13, 14, 15);
}
// B tile stored n-major in LDS: [16 n][32 k]; one aligned 32B vector load.
__device__ __forceinline__ v16bf frag_bt(const __bf16* Bt, int lane) {
  const __bf16* p = Bt + (lane & 15) * 32 + ((lane >> 4) << 4);
  return *(const v16bf*)(p);
}
__device__ __forceinline__ float sigmoidf_(float x) { return 1.0f / (1.0f + expf(-x)); }

// ---------------- prep kernels ----------------
// ua_w (k,n) f32 -> bf16 transposed [n][k]
__global__ void k_cvt_ua(const float* __restrict__ w, __bf16* __restrict__ o) {
  int i = blockIdx.x * 256 + threadIdx.x;            // 1024*256
  int n = i >> 10, k = i & 1023;
  o[i] = (__bf16)w[k * 256 + n];
}
// conv_tan_w (oc,ic,kh,kw) f32 -> bf16 [tap][oc][ic]
__global__ void k_cvt_convw(const float* __restrict__ w, __bf16* __restrict__ o) {
  int i = blockIdx.x * 256 + threadIdx.x;            // 9*256*256
  int tap = i / 65536;
  int rem = i & 65535;
  int oc = rem >> 8, ic = rem & 255;
  o[i] = (__bf16)w[oc * (256 * 9) + ic * 9 + tap];
}
__global__ void k_embed(const int* __restrict__ y, const float* __restrict__ tbl,
                        float* __restrict__ emb) {
  int i = blockIdx.x * 256 + threadIdx.x;            // 16*256
  int b = i >> 8, j = i & 255;
  emb[i] = tbl[y[b] * EMBN + j];
}
// conv1(alpha_t) + attention_sum -> att_sum output
__global__ void k_conv1_attsum(const float* __restrict__ alpha_t,
                               const float* __restrict__ att_in,
                               const float* __restrict__ w9, const float* __restrict__ b1,
                               float* __restrict__ att_out) {
  int m = blockIdx.x * 256 + threadIdx.x;            // 36864
  int b = m / HWn; int r = m - b * HWn; int h = r / Ww; int w = r - h * Ww;
  float acc = b1[0];
#pragma unroll
  for (int kh = 0; kh < 3; ++kh)
#pragma unroll
    for (int kw = 0; kw < 3; ++kw) {
      int hh = h + kh - 1, ww = w + kw - 1;
      if (hh >= 0 && hh < Hh && ww >= 0 && ww < Ww)
        acc += alpha_t[b * HWn + hh * Ww + ww] * w9[kh * 3 + kw];
    }
  att_out[m] = att_in[m] + acc;
}
// ---------------- GRU building blocks (small, scalar) ----------------
__global__ void k_gru_mm(const float* __restrict__ x, int kx, const float* __restrict__ wi,
                         const float* __restrict__ bi,
                         const float* __restrict__ h, int kh, const float* __restrict__ wh,
                         const float* __restrict__ bh,
                         float* __restrict__ gi, float* __restrict__ gh) {
  int i = blockIdx.x * 256 + threadIdx.x;            // 16*768
  int b = i / 768, j = i - b * 768;
  float a = bi[j];
  for (int k = 0; k < kx; ++k) a += x[b * kx + k] * wi[k * 768 + j];
  gi[i] = a;
  float c = bh[j];
  for (int k = 0; k < kh; ++k) c += h[b * kh + k] * wh[k * 768 + j];
  gh[i] = c;
}
__global__ void k_gru_gate(const float* __restrict__ gi, const float* __restrict__ gh,
                           const float* __restrict__ hprev, float* __restrict__ hout,
                           float* __restrict__ hout2) {
  int i = blockIdx.x * 256 + threadIdx.x;            // 16*256
  int b = i >> 8, n = i & 255;
  float ir = gi[b * 768 + n],        hr = gh[b * 768 + n];
  float iz = gi[b * 768 + 256 + n],  hz = gh[b * 768 + 256 + n];
  float in_ = gi[b * 768 + 512 + n], hn = gh[b * 768 + 512 + n];
  float rr = sigmoidf_(ir + hr);
  float zz = sigmoidf_(iz + hz);
  float nn = tanhf(in_ + rr * hn);
  float v = (1.0f - zz) * nn + zz * hprev[i];
  hout[i] = v;
  if (hout2) hout2[i] = v;
}
__global__ void k_fc1(const float* __restrict__ x, const float* __restrict__ w,
                      const float* __restrict__ b, float* __restrict__ o) {
  int i = blockIdx.x * 256 + threadIdx.x;            // 16*256
  int bb = i >> 8, n = i & 255;
  float a = b[n];
  for (int k = 0; k < 256; ++k) a += x[bb * 256 + k] * w[k * 256 + n];
  o[i] = a;
}

// ---------------- big GEMM: et = img^T @ ua_w (+fused adds), bf16 WMMA ----
// M tile 64, N 256, K 1024, double-buffered LDS, vectorized staging.
__device__ __forceinline__ void stage_et(const float* __restrict__ img,
                                         const __bf16* __restrict__ uabfT,
                                         int m0, int ks, __bf16* sA, __bf16* sBt,
                                         int tid) {
  // A: thread -> (row = tid&63, 8 consecutive k). 8 clause-able loads,
  // pack to bf16x8, single ds_store_b128.
  const int row = tid & 63, kk0 = (tid >> 6) << 3;
  const int m = m0 + row;
  const int b = m / HWn;
  const int r = m - b * HWn;
  const float* src = img + (b * Cc + ks * 32 + kk0) * HWn + r;
  float v[8];
#pragma unroll
  for (int i = 0; i < 8; ++i) v[i] = src[i * HWn];
  v8bf pk;
#pragma unroll
  for (int i = 0; i < 8; ++i) pk[i] = (__bf16)v[i];
  *(v8bf*)(sA + row * 32 + kk0) = pk;
  // B: 1024 16-byte chunks, 4 per thread, contiguous both sides.
#pragma unroll
  for (int i = 0; i < 4; ++i) {
    int c = i * 256 + tid;
    int n = c >> 2, kq = (c & 3) << 3;
    *(v8bf*)(sBt + c * 8) = *(const v8bf*)(uabfT + n * 1024 + ks * 32 + kq);
  }
}

__global__ __launch_bounds__(256) void k_et_gemm(
    const float* __restrict__ img, const __bf16* __restrict__ uabfT,
    const float* __restrict__ uab, const float* __restrict__ st1,
    const float* __restrict__ att_sum, const float* __restrict__ ufw,
    const float* __restrict__ ufb, __bf16* __restrict__ et) {
  __shared__ __align__(32) __bf16 sA[2][64 * 32];
  __shared__ __align__(32) __bf16 sBt[2][256 * 32];  // [n][k]
  const int tid = threadIdx.x, lane = tid & 31, wv = tid >> 5;
  const int m0 = blockIdx.x * 64;
  const int msub = wv >> 1, nhalf = wv & 1;
  v8f acc[8] = {};
  stage_et(img, uabfT, m0, 0, sA[0], sBt[0], tid);
  for (int ks = 0; ks < 32; ++ks) {
    __syncthreads();
    if (ks < 31) stage_et(img, uabfT, m0, ks + 1, sA[(ks + 1) & 1], sBt[(ks + 1) & 1], tid);
    const __bf16* cA = sA[ks & 1];
    const __bf16* cB = sBt[ks & 1];
    // preload all fragments, then issue WMMAs back-to-back (keeps accs pinned)
    v16bf a = frag_a(cA + msub * 16 * 32, lane);
    v16bf bfr[8];
#pragma unroll
    for (int j = 0; j < 8; ++j)
      bfr[j] = frag_bt(cB + (nhalf * 128 + j * 16) * 32, lane);
#pragma unroll
    for (int j = 0; j < 8; ++j)
      acc[j] = __builtin_amdgcn_wmma_f32_16x16x32_bf16(false, a, false, bfr[j],
                                                       (short)0, acc[j], false, false);
  }
#pragma unroll
  for (int r = 0; r < 8; ++r) {
    int ml = msub * 16 + r + ((lane >> 4) << 3);
    int m = m0 + ml;
    int b = m / HWn;
    float as = att_sum[m];
#pragma unroll
    for (int j = 0; j < 8; ++j) {
      int n = nhalf * 128 + j * 16 + (lane & 15);
      float v = acc[j][r] + uab[n] + st1[b * 256 + n] + as * ufw[n] + ufb[n];
      et[m * 256 + n] = (__bf16)v;
    }
  }
}

// ------- implicit-GEMM 3x3 conv 256->256 + mask/BN/tanh/v-dot -> e_exp ----
// one block per (b,h): M=96 (all w), N=256, K=9*256. k-chunk outer loop
// stages a zero-padded 3x98x32 halo once (vector loads); taps double-buffer B.
__device__ __forceinline__ void stage_bt_conv(const __bf16* __restrict__ wT,
                                              int tap, int ks, __bf16* sBt, int tid) {
#pragma unroll
  for (int i = 0; i < 4; ++i) {
    int c = i * 256 + tid;
    int n = c >> 2, kq = (c & 3) << 3;
    *(v8bf*)(sBt + c * 8) = *(const v8bf*)(wT + (tap * 256 + n) * 256 + ks * 32 + kq);
  }
}

__global__ __launch_bounds__(256) void k_conv_tan(
    const __bf16* __restrict__ et, const __bf16* __restrict__ wT,
    const float* __restrict__ cb, const int* __restrict__ hmask,
    const int* __restrict__ wmask, const float* __restrict__ bng,
    const float* __restrict__ bnb, const float* __restrict__ bnm,
    const float* __restrict__ bnv, const float* __restrict__ vw,
    const float* __restrict__ vb, float* __restrict__ e_exp) {
  __shared__ __align__(32) __bf16 sH[3 * 98 * 32];   // [hsel][w+1][k]
  __shared__ __align__(32) __bf16 sBt[2][256 * 32];  // [oc][k], ping-pong
  __shared__ float sE[96];
  const int tid = threadIdx.x, lane = tid & 31, wv = tid >> 5;
  const int b = blockIdx.x / Hh, h = blockIdx.x - b * Hh;
  if (tid < 96) sE[tid] = 0.0f;
  v8f acc[6][2] = {};
  for (int ks = 0; ks < 8; ++ks) {
    __syncthreads();                                  // prior tap reads done
    for (int cv = tid; cv < 3 * 98 * 4; cv += 256) {  // halo, 16B vector moves
      int hs = cv / (98 * 4);
      int r  = cv - hs * 98 * 4;
      int wp = r >> 2, kq = (r & 3) << 3;
      int hh = h + hs - 1, w = wp - 1;
      v8bf val = {};
      if (hh >= 0 && hh < Hh && w >= 0 && w < Ww)
        val = *(const v8bf*)(et + ((b * Hh + hh) * Ww + w) * 256 + ks * 32 + kq);
      *(v8bf*)(sH + cv * 8) = val;
    }
    stage_bt_conv(wT, 0, ks, sBt[0], tid);
#pragma unroll
    for (int tap = 0; tap < 9; ++tap) {
      const int dh = tap / 3 - 1, dw = tap % 3 - 1;
      __syncthreads();
      if (tap < 8) stage_bt_conv(wT, tap + 1, ks, sBt[(tap + 1) & 1], tid);
      const __bf16* cB = sBt[tap & 1];
      // preload all fragments, then issue WMMAs back-to-back
      v16bf b0 = frag_bt(cB + (wv * 32) * 32, lane);
      v16bf b1 = frag_bt(cB + (wv * 32 + 16) * 32, lane);
      v16bf afr[6];
#pragma unroll
      for (int ms = 0; ms < 6; ++ms)
        afr[ms] = frag_a(sH + ((dh + 1) * 98 + ms * 16 + dw + 1) * 32, lane);
#pragma unroll
      for (int ms = 0; ms < 6; ++ms) {
        acc[ms][0] = __builtin_amdgcn_wmma_f32_16x16x32_bf16(false, afr[ms], false, b0,
                                                             (short)0, acc[ms][0], false, false);
        acc[ms][1] = __builtin_amdgcn_wmma_f32_16x16x32_bf16(false, afr[ms], false, b1,
                                                             (short)0, acc[ms][1], false, false);
      }
    }
  }
  __syncthreads();
  const int hm = hmask[b], wm = wmask[b];
#pragma unroll
  for (int ms = 0; ms < 6; ++ms)
#pragma unroll
    for (int t = 0; t < 2; ++t)
#pragma unroll
      for (int r = 0; r < 8; ++r) {
        int ml = ms * 16 + r + ((lane >> 4) << 3);   // output w position
        int n  = wv * 32 + t * 16 + (lane & 15);     // output channel
        float v = acc[ms][t][r] + cb[n];
        float msk = (h < hm && ml < wm) ? 1.0f : 0.0f;
        v *= msk;
        float inv = bng[n] * rsqrtf(bnv[n] + BN_EPS);
        v = (v - bnm[n]) * inv + bnb[n];
        atomicAdd(&sE[ml], tanhf(v) * vw[n]);
      }
  __syncthreads();
  if (tid < 96) {
    float e = sE[tid] + vb[0];
    float msk = (h < hm && tid < wm) ? 1.0f : 0.0f;
    e_exp[(b * Hh + h) * Ww + tid] = expf(e) * msk;
  }
}

// ---------------- softmax / context ----------------
__global__ void k_esum(const float* __restrict__ e_exp, float* __restrict__ esum) {
  __shared__ float s[256];
  int b = blockIdx.x;
  float a = 0.0f;
  for (int i = threadIdx.x; i < HWn; i += 256) a += e_exp[b * HWn + i];
  s[threadIdx.x] = a; __syncthreads();
  for (int st = 128; st > 0; st >>= 1) {
    if (threadIdx.x < st) s[threadIdx.x] += s[threadIdx.x + st];
    __syncthreads();
  }
  if (threadIdx.x == 0) esum[b] = s[0];
}
__global__ void k_alpha(const float* __restrict__ e_exp, const float* __restrict__ esum,
                        float* __restrict__ alpha) {
  int m = blockIdx.x * 256 + threadIdx.x;            // 36864
  int b = m / HWn;
  alpha[m] = e_exp[m] / (esum[b] + 1e-8f);
}
__global__ void k_ct(const float* __restrict__ alpha, const float* __restrict__ img,
                     float* __restrict__ ct) {
  int b = blockIdx.x >> 2;
  int c = (blockIdx.x & 3) * 256 + threadIdx.x;
  const float* ip = img + (b * Cc + c) * HWn;
  const float* ap = alpha + b * HWn;
  float a = 0.0f;
  for (int i = 0; i < HWn; ++i) {
    if ((i & 255) == 0) __builtin_prefetch(ip + i + 256, 0, 1);
    a += ap[i] * ip[i];
  }
  ct[b * Cc + c] = a;
}

// ---------------- output head ----------------
__global__ void k_pre(const float* __restrict__ st, const float* __restrict__ fc2w,
                      const float* __restrict__ fc2b, const float* __restrict__ emb,
                      const float* __restrict__ e2w, const float* __restrict__ e2b,
                      const float* __restrict__ ct, const float* __restrict__ wcw,
                      const float* __restrict__ wcb, float* __restrict__ pre) {
  int i = blockIdx.x * 256 + threadIdx.x;            // 16*128
  if (i >= Bq * 128) return;
  int b = i >> 7, j = i & 127;
  float a = fc2b[j] + e2b[j] + wcb[j];
  for (int k = 0; k < 256; ++k) a += st[b * 256 + k] * fc2w[k * 128 + j];
  for (int k = 0; k < 256; ++k) a += emb[b * 256 + k] * e2w[k * 128 + j];
  for (int k = 0; k < 1024; ++k) a += ct[b * 1024 + k] * wcw[k * 128 + j];
  pre[i] = a;
}
__global__ void k_logits(const float* __restrict__ pre, const float* __restrict__ ow,
                         const float* __restrict__ ob, float* __restrict__ lg) {
  int i = blockIdx.x * 256 + threadIdx.x;            // 16*5000
  if (i >= Bq * Vv) return;
  int b = i / Vv, v = i - b * Vv;
  float a = ob[v];
  for (int k = 0; k < 128; ++k) a += pre[b * 128 + k] * ow[k * Vv + v];
  lg[i] = a;
}
__global__ void k_logsoftmax(const float* __restrict__ lg, float* __restrict__ out) {
  __shared__ float s[256];
  int b = blockIdx.x;
  const float* x = lg + b * Vv;
  float m = -1e30f;
  for (int i = threadIdx.x; i < Vv; i += 256) m = fmaxf(m, x[i]);
  s[threadIdx.x] = m; __syncthreads();
  for (int st = 128; st > 0; st >>= 1) {
    if (threadIdx.x < st) s[threadIdx.x] = fmaxf(s[threadIdx.x], s[threadIdx.x + st]);
    __syncthreads();
  }
  float mx = s[0]; __syncthreads();
  float a = 0.0f;
  for (int i = threadIdx.x; i < Vv; i += 256) a += expf(x[i] - mx);
  s[threadIdx.x] = a; __syncthreads();
  for (int st = 128; st > 0; st >>= 1) {
    if (threadIdx.x < st) s[threadIdx.x] += s[threadIdx.x + st];
    __syncthreads();
  }
  float lse = mx + logf(s[0]);
  for (int i = threadIdx.x; i < Vv; i += 256) out[b * Vv + i] = x[i] - lse;
}

// ---------------- launch ----------------
extern "C" void kernel_launch(void* const* d_in, const int* in_sizes, int n_in,
                              void* d_out, int out_size, void* d_ws, size_t ws_size,
                              hipStream_t stream) {
  (void)in_sizes; (void)n_in; (void)out_size; (void)ws_size;
  const int*   input_y  = (const int*)d_in[0];
  const float* hid0     = (const float*)d_in[1];
  const float* img      = (const float*)d_in[2];
  const float* att_in   = (const float*)d_in[4];
  const float* alpha_t  = (const float*)d_in[5];
  const int*   h_mask   = (const int*)d_in[8];
  const int*   w_mask   = (const int*)d_in[9];
  const float* emb_tbl  = (const float*)d_in[10];
  const float* g1wi = (const float*)d_in[11]; const float* g1wh = (const float*)d_in[12];
  const float* g1bi = (const float*)d_in[13]; const float* g1bh = (const float*)d_in[14];
  const float* fc1w = (const float*)d_in[15]; const float* fc1b = (const float*)d_in[16];
  const float* g2wi = (const float*)d_in[17]; const float* g2wh = (const float*)d_in[18];
  const float* g2bi = (const float*)d_in[19]; const float* g2bh = (const float*)d_in[20];
  const float* outw = (const float*)d_in[21]; const float* outb = (const float*)d_in[22];
  const float* e2w  = (const float*)d_in[23]; const float* e2b  = (const float*)d_in[24];
  const float* c1w  = (const float*)d_in[25]; const float* c1b  = (const float*)d_in[26];
  const float* ctw  = (const float*)d_in[27]; const float* ctb  = (const float*)d_in[28];
  const float* fc2w = (const float*)d_in[29]; const float* fc2b = (const float*)d_in[30];
  const float* uaw  = (const float*)d_in[31]; const float* uab  = (const float*)d_in[32];
  const float* ufw  = (const float*)d_in[33]; const float* ufb  = (const float*)d_in[34];
  const float* vw   = (const float*)d_in[35]; const float* vb   = (const float*)d_in[36];
  const float* wcw  = (const float*)d_in[37]; const float* wcb  = (const float*)d_in[38];
  const float* bng  = (const float*)d_in[39]; const float* bnb  = (const float*)d_in[40];
  const float* bnm  = (const float*)d_in[41]; const float* bnv  = (const float*)d_in[42];

  float* out0 = (float*)d_out;                 // output   [16,5000]
  float* out1 = out0 + Bq * Vv;                // hidden   [16,256]
  float* out2 = out1 + Bq * HIDN;              // alpha    [16,24,96]
  float* out3 = out2 + BHWn;                   // att_sum  [16,24,96]

  char* ws = (char*)d_ws;
  float*  w_emb   = (float*)(ws + 0);          // 16*256
  float*  w_st1g  = (float*)(ws + 16384);      // gru1 raw out
  float*  w_st1   = (float*)(ws + 32768);      // post-fc1
  float*  w_gi1   = (float*)(ws + 49152);      // 16*768
  float*  w_gh1   = (float*)(ws + 98304);
  __bf16* w_uabfT = (__bf16*)(ws + 147456);    // 256n x 1024k bf16
  __bf16* w_wT    = (__bf16*)(ws + 671744);    // 9 x 256oc x 256ic bf16
  __bf16* w_et    = (__bf16*)(ws + 1851392);   // 36864*256 bf16
  float*  w_eexp  = (float*)(ws + 20725760);   // 36864
  float*  w_esum  = (float*)(ws + 20873216);   // 16
  float*  w_ct    = (float*)(ws + 20873280);   // 16*1024
  float*  w_gi2   = (float*)(ws + 20938816);
  float*  w_gh2   = (float*)(ws + 20987968);
  float*  w_st    = (float*)(ws + 21037120);   // 16*256
  float*  w_pre   = (float*)(ws + 21053504);   // 16*128
  float*  w_lg    = (float*)(ws + 21061696);   // 16*5000

  // prep
  k_cvt_ua<<<1024, 256, 0, stream>>>(uaw, w_uabfT);
  k_cvt_convw<<<2304, 256, 0, stream>>>(ctw, w_wT);
  k_embed<<<16, 256, 0, stream>>>(input_y, emb_tbl, w_emb);
  k_conv1_attsum<<<144, 256, 0, stream>>>(alpha_t, att_in, c1w, c1b, out3);
  // GRU1 + fc1
  k_gru_mm<<<48, 256, 0, stream>>>(w_emb, 256, g1wi, g1bi, hid0, 256, g1wh, g1bh, w_gi1, w_gh1);
  k_gru_gate<<<16, 256, 0, stream>>>(w_gi1, w_gh1, hid0, w_st1g, (float*)nullptr);
  k_fc1<<<16, 256, 0, stream>>>(w_st1g, fc1w, fc1b, w_st1);
  // big fused einsum -> et (bf16 NHWC)
  k_et_gemm<<<576, 256, 0, stream>>>(img, w_uabfT, uab, w_st1, out3, ufw, ufb, w_et);
  // conv_tan + mask/BN/tanh/v-dot -> e_exp
  k_conv_tan<<<Bq * Hh, 256, 0, stream>>>(w_et, w_wT, ctb, h_mask, w_mask,
                                          bng, bnb, bnm, bnv, vw, vb, w_eexp);
  // softmax + context
  k_esum<<<16, 256, 0, stream>>>(w_eexp, w_esum);
  k_alpha<<<144, 256, 0, stream>>>(w_eexp, w_esum, out2);
  k_ct<<<64, 256, 0, stream>>>(out2, img, w_ct);
  // GRU2
  k_gru_mm<<<48, 256, 0, stream>>>(w_ct, 1024, g2wi, g2bi, w_st1, 256, g2wh, g2bh, w_gi2, w_gh2);
  k_gru_gate<<<16, 256, 0, stream>>>(w_gi2, w_gh2, w_st1, w_st, out1);
  // head
  k_pre<<<8, 256, 0, stream>>>(w_st, fc2w, fc2b, w_emb, e2w, e2b, w_ct, wcw, wcb, w_pre);
  k_logits<<<(Bq * Vv + 255) / 256, 256, 0, stream>>>(w_pre, outw, outb, w_lg);
  k_logsoftmax<<<16, 256, 0, stream>>>(w_lg, out0);
}
